// MixedAttentionBlock_80290118632213
// MI455X (gfx1250) — compile-verified, matrix operand order checked by reading, not run on hardware
//
#include <hip/hip_runtime.h>
#include <hip/hip_bf16.h>

// ---------------------------------------------------------------------------
// MixedAttentionBlock for MI455X (gfx1250): bf16 WMMA GEMMs + fused epilogues
// B=32 H=W=64 C=256 NH=8 WS=8 SS=4 -> N=64 tok/win, NW=64 win/img, HD=32
// GEMM v3: 128x64 block tile, 32x32 wave tiles, DOUBLE-BUFFERED async
//          global->LDS tile loads (ASYNCcnt software pipeline)
// ---------------------------------------------------------------------------

#define BB   32
#define HH   64
#define WW   64
#define CC   256
#define NHD  8
#define WSZ  8
#define SSH  4
#define NTOK (WSZ * WSZ)          // 64 tokens per window
#define NWIN ((HH / WSZ) * (WW / WSZ)) // 64 windows per image
#define HDIM (CC / NHD)           // 32
#define MTOT (BB * HH * WW)       // 131072 tokens
#define MLPH (4 * CC)             // 1024
#define CABH (CC / 16)            // 16
#define EPSV 1e-5f

typedef __attribute__((ext_vector_type(16))) __bf16 v16bf;
typedef __attribute__((ext_vector_type(8)))  float  v8f;

union FragBF { v16bf v; uint4 u[2]; };

static __device__ __forceinline__ v8f wmma_bf16(v16bf a, v16bf b, v8f c) {
    return __builtin_amdgcn_wmma_f32_16x16x32_bf16(false, a, false, b, (short)0, c, false, false);
}

// CDNA5 async copy: memory -> LDS (tracked with ASYNCcnt).
// %0 = per-lane LDS byte offset, %1 = per-lane 64-bit global address.
static __device__ __forceinline__ void async_ld_b128(const void* g, void* l) {
    unsigned lds_off = (unsigned)(uintptr_t)l;
    unsigned long long ga = (unsigned long long)(uintptr_t)g;
    asm volatile("global_load_async_to_lds_b128 %0, %1, off"
                 :: "v"(lds_off), "v"(ga)
                 : "memory");
}
static __device__ __forceinline__ void wait_async0() {
    asm volatile("s_wait_asynccnt 0x0" ::: "memory");
}
// Allow the newest issue group (3 async loads) to remain in flight; async
// loads complete in order, so <=3 means the older tile's group has landed.
static __device__ __forceinline__ void wait_async3() {
    asm volatile("s_wait_asynccnt 0x3" ::: "memory");
}

// ------------------------------- fp32 -> bf16 weight convert ----------------
__global__ __launch_bounds__(256) void f2bf_kernel(const float* __restrict__ s,
                                                   __bf16* __restrict__ d, int n) {
    int i = blockIdx.x * 256 + threadIdx.x;
    if (i < n) d[i] = (__bf16)s[i];
}

// ------------------------------- LayerNorm (wave per token) -----------------
// PERM: apply cyclic shift + window partition (output in window-token order)
// OUTBF: write bf16 (else fp32)
template<bool PERM, bool OUTBF>
__global__ __launch_bounds__(256) void ln_kernel(const float* __restrict__ x,
                                                 const float* __restrict__ g,
                                                 const float* __restrict__ bta,
                                                 float* __restrict__ outf,
                                                 __bf16* __restrict__ outb) {
    int wave = threadIdx.x >> 5, lane = threadIdx.x & 31;
    int t = blockIdx.x * 8 + wave;         // destination token (window order if PERM)
    int src;
    if (PERM) {
        int win = t >> 6, n = t & 63;
        int bi = win >> 6, iw = win & 63;
        int wy = iw >> 3, wx = iw & 7, iy = n >> 3, ix = n & 7;
        int sy = ((wy << 3) + iy + SSH) & (HH - 1);
        int sx = ((wx << 3) + ix + SSH) & (WW - 1);
        src = (bi << 12) + (sy << 6) + sx;
    } else {
        src = t;
    }
    const float4* p = (const float4*)(x + (size_t)src * CC) + lane * 2;
    float4 a0 = p[0], a1 = p[1];
    float v[8] = {a0.x, a0.y, a0.z, a0.w, a1.x, a1.y, a1.z, a1.w};
    float s = 0.f;
    #pragma unroll
    for (int j = 0; j < 8; ++j) s += v[j];
    #pragma unroll
    for (int o = 16; o > 0; o >>= 1) s += __shfl_xor(s, o);
    float mean = s * (1.f / 256.f);
    float q = 0.f;
    #pragma unroll
    for (int j = 0; j < 8; ++j) { float d = v[j] - mean; q += d * d; }
    #pragma unroll
    for (int o = 16; o > 0; o >>= 1) q += __shfl_xor(q, o);
    float rstd = rsqrtf(q * (1.f / 256.f) + EPSV);
    int cb = lane * 8;
    if (OUTBF) {
        union { __bf16 h[8]; uint4 u; } ob;
        #pragma unroll
        for (int j = 0; j < 8; ++j)
            ob.h[j] = (__bf16)((v[j] - mean) * rstd * g[cb + j] + bta[cb + j]);
        *(uint4*)(outb + (size_t)t * CC + cb) = ob.u;
    } else {
        float4 r0, r1;
        r0.x = (v[0]-mean)*rstd*g[cb+0]+bta[cb+0]; r0.y = (v[1]-mean)*rstd*g[cb+1]+bta[cb+1];
        r0.z = (v[2]-mean)*rstd*g[cb+2]+bta[cb+2]; r0.w = (v[3]-mean)*rstd*g[cb+3]+bta[cb+3];
        r1.x = (v[4]-mean)*rstd*g[cb+4]+bta[cb+4]; r1.y = (v[5]-mean)*rstd*g[cb+5]+bta[cb+5];
        r1.z = (v[6]-mean)*rstd*g[cb+6]+bta[cb+6]; r1.w = (v[7]-mean)*rstd*g[cb+7]+bta[cb+7];
        float4* op = (float4*)(outf + (size_t)t * CC) + lane * 2;
        op[0] = r0; op[1] = r1;
    }
}

// ------------------------------- generic WMMA GEMM --------------------------
// out[M,N] = epilogue(A[M,K](bf16) @ W[N,K]^T(bf16) + bias[N])
// block tile 128x64, 8 waves as 4(M)x2(N), each wave 32x32 -> 4 wmma / K-step.
// Double-buffered GLOBAL_LOAD_ASYNC_TO_LDS_B128 pipeline: tile k+1 streams
// into LDS while tile k feeds WMMA.
// MODE 0: store bf16            MODE 1: exact-GELU, store bf16
// MODE 2: +res[m], store fp32   MODE 3: window->spatial permute, +res, fp32
template<int MODE>
__global__ __launch_bounds__(256) void gemm_kernel(const __bf16* __restrict__ A,
                                                   const __bf16* __restrict__ Wt,
                                                   const float* __restrict__ bias,
                                                   const float* __restrict__ res,
                                                   float* __restrict__ outf,
                                                   __bf16* __restrict__ outb,
                                                   int M, int N, int K) {
    int nb = N >> 6;
    int bm = blockIdx.x / nb, bn = blockIdx.x % nb;
    int tid = threadIdx.x;
    int wave = tid >> 5, lane = tid & 31;
    int wm = wave >> 1, wn = wave & 1;
    int hb = lane >> 4, l16 = lane & 15;
    __shared__ alignas(16) __bf16 As[2][128][40];
    __shared__ alignas(16) __bf16 Bs[2][64][40];
    v8f acc[2][2];
    #pragma unroll
    for (int tm = 0; tm < 2; ++tm)
        #pragma unroll
        for (int tn = 0; tn < 2; ++tn) { v8f z = {0,0,0,0,0,0,0,0}; acc[tm][tn] = z; }
    int m0 = bm << 7, n0 = bn << 6;
    // per-thread tile chunks: A 128x32 = 512 x 16B (2/thread), B 64x32 = 256 (1/thread)
    int ar0 = tid >> 2,          ac0 = (tid & 3) * 8;
    int ar1 = (tid + 256) >> 2,  ac1 = ac0;  // (tid+256)&3 == tid&3
    int brw = tid >> 2,          bcw = (tid & 3) * 8;
    const __bf16* Ap0 = A  + (size_t)(m0 + ar0) * K + ac0;
    const __bf16* Ap1 = A  + (size_t)(m0 + ar1) * K + ac1;
    const __bf16* Bpw = Wt + (size_t)(n0 + brw) * K + bcw;
    int nsteps = K >> 5;
    // prologue: stream tile 0 into buffer 0
    async_ld_b128(Ap0, &As[0][ar0][ac0]);
    async_ld_b128(Ap1, &As[0][ar1][ac1]);
    async_ld_b128(Bpw, &Bs[0][brw][bcw]);
    for (int k = 0; k < nsteps; ++k) {
        int cur = k & 1, nxt = cur ^ 1;
        // guards buffer 'nxt' (last read in the previous iteration) before overwrite
        __syncthreads();
        if (k + 1 < nsteps) {
            int kb = (k + 1) << 5;
            async_ld_b128(Ap0 + kb, &As[nxt][ar0][ac0]);
            async_ld_b128(Ap1 + kb, &As[nxt][ar1][ac1]);
            async_ld_b128(Bpw + kb, &Bs[nxt][brw][bcw]);
            wait_async3();      // drain the older (cur) group only
        } else {
            wait_async0();
        }
        __syncthreads();        // all waves' 'cur' tile landed in LDS
        FragBF af[2], bf[2];
        #pragma unroll
        for (int tm = 0; tm < 2; ++tm) {
            int ar = (wm << 5) + (tm << 4) + l16;
            af[tm].u[0] = *(const uint4*)(&As[cur][ar][hb * 8]);
            af[tm].u[1] = *(const uint4*)(&As[cur][ar][16 + hb * 8]);
        }
        #pragma unroll
        for (int tn = 0; tn < 2; ++tn) {
            int br = (wn << 5) + (tn << 4) + l16;
            bf[tn].u[0] = *(const uint4*)(&Bs[cur][br][hb * 8]);
            bf[tn].u[1] = *(const uint4*)(&Bs[cur][br][16 + hb * 8]);
        }
        #pragma unroll
        for (int tm = 0; tm < 2; ++tm)
            #pragma unroll
            for (int tn = 0; tn < 2; ++tn)
                acc[tm][tn] = wmma_bf16(af[tm].v, bf[tn].v, acc[tm][tn]);
    }
    #pragma unroll
    for (int tm = 0; tm < 2; ++tm) {
        #pragma unroll
        for (int tn = 0; tn < 2; ++tn) {
            int n = n0 + (wn << 5) + (tn << 4) + l16;
            float bn_ = bias[n];
            #pragma unroll
            for (int v = 0; v < 8; ++v) {
                int m = m0 + (wm << 5) + (tm << 4) + v + (hb << 3);
                float val = acc[tm][tn][v] + bn_;
                if (MODE == 1) val = 0.5f * val * (1.f + erff(val * 0.70710678118654752f));
                if (MODE == 0 || MODE == 1) {
                    outb[(size_t)m * N + n] = (__bf16)val;
                } else if (MODE == 2) {
                    size_t idx = (size_t)m * N + n;
                    outf[idx] = val + res[idx];
                } else { // MODE 3: window reverse + roll(+SS)
                    int win = m >> 6, nt = m & 63;
                    int bi = win >> 6, iw = win & 63;
                    int wy = iw >> 3, wx = iw & 7, iy = nt >> 3, ix = nt & 7;
                    int sy = ((wy << 3) + iy + SSH) & (HH - 1);
                    int sx = ((wx << 3) + ix + SSH) & (WW - 1);
                    size_t sp = ((size_t)bi << 12) + (sy << 6) + sx;
                    size_t idx = sp * N + n;
                    outf[idx] = val + res[idx];
                }
            }
        }
    }
}

// ------------------------------- windowed attention -------------------------
// block: 256 thr = 8 waves; 4 heads/block, 2 waves/head (32 rows each)
__global__ __launch_bounds__(256) void attn_kernel(const __bf16* __restrict__ qkv,
                                                   const float* __restrict__ rpb,
                                                   __bf16* __restrict__ out) {
    int blk = blockIdx.x;
    int win = blk >> 1, hg = blk & 1;
    int tid = threadIdx.x, wave = tid >> 5, lane = tid & 31;
    int hl = wave >> 1;                 // local head 0..3
    int h  = (hg << 2) + hl;
    int msub = wave & 1;                // row half of the head
    int hb = lane >> 4, l16 = lane & 15;

    __shared__ float rpb_s[225 * NHD];
    __shared__ int regid[64];
    __shared__ alignas(16) __bf16 Pb[4][64][72];
    __shared__ alignas(16) __bf16 Vt[4][32][72];

    for (int i = tid; i < 225 * NHD; i += 256) rpb_s[i] = rpb[i];
    if (tid < 64) {
        int iw = win & 63, wy = iw >> 3, wx = iw & 7;
        int iy = tid >> 3, ix = tid & 7;
        int sy = wy * 8 + iy, sx = wx * 8 + ix;
        int ry = sy < (HH - WSZ) ? 0 : (sy < (HH - SSH) ? 1 : 2);
        int rx = sx < (WW - WSZ) ? 0 : (sx < (WW - SSH) ? 1 : 2);
        regid[tid] = ry * 3 + rx;
    }
    {   // stage V transposed: Vt[d][m] = V[m][d]
        int hv = tid >> 6, m = tid & 63;
        const __bf16* vp = qkv + ((size_t)(win * 64 + m)) * (3 * CC) + 2 * CC + ((hg << 2) + hv) * HDIM;
        #pragma unroll
        for (int j = 0; j < HDIM; ++j) Vt[hv][j][m] = vp[j];
    }
    __syncthreads();

    size_t rowbase = (size_t)win * 64;
    FragBF qa[2], kb[4];
    #pragma unroll
    for (int mi = 0; mi < 2; ++mi) {
        int r = msub * 32 + mi * 16 + l16;
        const __bf16* p = qkv + (rowbase + r) * (3 * CC) + h * HDIM;
        qa[mi].u[0] = *(const uint4*)(p + hb * 8);
        qa[mi].u[1] = *(const uint4*)(p + 16 + hb * 8);
    }
    #pragma unroll
    for (int nj = 0; nj < 4; ++nj) {
        int r = nj * 16 + l16;
        const __bf16* p = qkv + (rowbase + r) * (3 * CC) + CC + h * HDIM;
        kb[nj].u[0] = *(const uint4*)(p + hb * 8);
        kb[nj].u[1] = *(const uint4*)(p + 16 + hb * 8);
    }

    v8f S[2][4];
    #pragma unroll
    for (int mi = 0; mi < 2; ++mi)
        #pragma unroll
        for (int nj = 0; nj < 4; ++nj) {
            v8f z = {0,0,0,0,0,0,0,0};
            S[mi][nj] = wmma_bf16(qa[mi].v, kb[nj].v, z);
        }

    const float scale = 0.17677669529663687f; // 1/sqrt(32)
    // bias + mask, then row softmax; rows live at (mi, v, hb) across nj tiles + 16 lanes
    #pragma unroll
    for (int mi = 0; mi < 2; ++mi) {
        #pragma unroll
        for (int v = 0; v < 8; ++v) {
            int qn = msub * 32 + mi * 16 + v + (hb << 3);
            int qiy = qn >> 3, qix = qn & 7, idq = regid[qn];
            float rowv[4];
            #pragma unroll
            for (int nj = 0; nj < 4; ++nj) {
                int km = nj * 16 + l16;
                int jy = km >> 3, jx = km & 7;
                int rpi = (qiy - jy + WSZ - 1) * (2 * WSZ - 1) + (qix - jx + WSZ - 1);
                float msk = (idq != regid[km]) ? -100.f : 0.f;
                rowv[nj] = S[mi][nj][v] * scale + rpb_s[rpi * NHD + h] + msk;
            }
            float rm = fmaxf(fmaxf(rowv[0], rowv[1]), fmaxf(rowv[2], rowv[3]));
            #pragma unroll
            for (int o = 8; o > 0; o >>= 1) rm = fmaxf(rm, __shfl_xor(rm, o));
            float rs = 0.f;
            #pragma unroll
            for (int nj = 0; nj < 4; ++nj) { rowv[nj] = __expf(rowv[nj] - rm); rs += rowv[nj]; }
            #pragma unroll
            for (int o = 8; o > 0; o >>= 1) rs += __shfl_xor(rs, o);
            float inv = 1.f / rs;
            #pragma unroll
            for (int nj = 0; nj < 4; ++nj)
                Pb[hl][qn][nj * 16 + l16] = (__bf16)(rowv[nj] * inv);
        }
    }
    __syncthreads();

    // O = P @ V  (K = 64 -> 2 wmma steps)
    v8f O[2][2];
    #pragma unroll
    for (int mi = 0; mi < 2; ++mi)
        #pragma unroll
        for (int nj = 0; nj < 2; ++nj) { v8f z = {0,0,0,0,0,0,0,0}; O[mi][nj] = z; }
    #pragma unroll
    for (int ks = 0; ks < 2; ++ks) {
        int kb0 = ks * 32;
        FragBF pf[2], vf[2];
        #pragma unroll
        for (int mi = 0; mi < 2; ++mi) {
            int r = msub * 32 + mi * 16 + l16;
            pf[mi].u[0] = *(const uint4*)(&Pb[hl][r][kb0 + hb * 8]);
            pf[mi].u[1] = *(const uint4*)(&Pb[hl][r][kb0 + 16 + hb * 8]);
        }
        #pragma unroll
        for (int nj = 0; nj < 2; ++nj) {
            int r = nj * 16 + l16;
            vf[nj].u[0] = *(const uint4*)(&Vt[hl][r][kb0 + hb * 8]);
            vf[nj].u[1] = *(const uint4*)(&Vt[hl][r][kb0 + 16 + hb * 8]);
        }
        #pragma unroll
        for (int mi = 0; mi < 2; ++mi)
            #pragma unroll
            for (int nj = 0; nj < 2; ++nj)
                O[mi][nj] = wmma_bf16(pf[mi].v, vf[nj].v, O[mi][nj]);
    }
    #pragma unroll
    for (int mi = 0; mi < 2; ++mi)
        #pragma unroll
        for (int nj = 0; nj < 2; ++nj)
            #pragma unroll
            for (int v = 0; v < 8; ++v) {
                int qn = msub * 32 + mi * 16 + v + (hb << 3);
                int d = nj * 16 + l16;
                out[(rowbase + qn) * CC + h * HDIM + d] = (__bf16)O[mi][nj][v];
            }
}

// ------------------------------- CAB: GAP/GMP reduce ------------------------
__global__ __launch_bounds__(256) void gapgmp_kernel(const float* __restrict__ yln,
                                                     float* __restrict__ gap,
                                                     float* __restrict__ gmp) {
    int b = blockIdx.x, c = threadIdx.x;
    const float* p = yln + ((size_t)b * (HH * WW)) * CC + c;
    float s = 0.f, mx = -3.4e38f;
    for (int i = 0; i < HH * WW; ++i) {
        float v = p[(size_t)i * CC];
        s += v; mx = fmaxf(mx, v);
    }
    gap[b * CC + c] = s * (1.f / (HH * WW));
    gmp[b * CC + c] = mx;
}

// ------------------------------- CAB: SE MLP --------------------------------
__global__ __launch_bounds__(256) void se_kernel(const float* __restrict__ gap,
                                                 const float* __restrict__ gmp,
                                                 const float* __restrict__ w1,
                                                 const float* __restrict__ b1,
                                                 const float* __restrict__ w2,
                                                 const float* __restrict__ b2,
                                                 float* __restrict__ ca) {
    int b = blockIdx.x, c = threadIdx.x;
    __shared__ float sg[CC], sm[CC], hg[CABH], hm[CABH];
    sg[c] = gap[b * CC + c];
    sm[c] = gmp[b * CC + c];
    __syncthreads();
    if (c < CABH) {
        float ag = 0.f, am = 0.f;
        for (int j = 0; j < CC; ++j) { ag += w1[c * CC + j] * sg[j]; am += w1[c * CC + j] * sm[j]; }
        hg[c] = fmaxf(ag + b1[c], 0.f);
        hm[c] = fmaxf(am + b1[c], 0.f);
    }
    __syncthreads();
    float s1 = b2[c], s2 = b2[c];
    #pragma unroll
    for (int j = 0; j < CABH; ++j) { s1 += w2[c * CABH + j] * hg[j]; s2 += w2[c * CABH + j] * hm[j]; }
    float vv = s1 + s2;
    ca[b * CC + c] = 1.f / (1.f + __expf(-vv));
}

// ----------------- CAB scale + depthwise 3x3 + residual + LN2 ---------------
__global__ __launch_bounds__(256) void cab_fuse_kernel(const float* __restrict__ x1,
                                                       const float* __restrict__ yln,
                                                       const float* __restrict__ ca,
                                                       const float* __restrict__ dww,
                                                       const float* __restrict__ dwb,
                                                       const float* __restrict__ n2g,
                                                       const float* __restrict__ n2b,
                                                       float* __restrict__ x2,
                                                       __bf16* __restrict__ hout) {
    int t = blockIdx.x, c = threadIdx.x;
    int b = t >> 12, p = t & 4095, yy = p >> 6, xx = p & 63;
    float cac = ca[(b << 8) + c];
    const float* base = yln + ((size_t)(b << 12)) * CC;
    float ctr = yln[(size_t)t * CC + c] * cac;
    float conv = 0.f;
    #pragma unroll
    for (int ky = 0; ky < 3; ++ky)
        #pragma unroll
        for (int kx = 0; kx < 3; ++kx) {
            int ny = yy + ky - 1, nx = xx + kx - 1;
            if (ny >= 0 && ny < HH && nx >= 0 && nx < WW)
                conv += dww[c * 9 + ky * 3 + kx] * base[(size_t)((ny << 6) + nx) * CC + c];
        }
    conv *= cac;
    float cab = ctr + conv + dwb[c];
    float xv = x1[(size_t)t * CC + c] + cab;
    __shared__ float red[CC];
    red[c] = xv; __syncthreads();
    for (int st = 128; st > 0; st >>= 1) { if (c < st) red[c] += red[c + st]; __syncthreads(); }
    float mean = red[0] * (1.f / CC);
    __syncthreads();
    float d = xv - mean;
    red[c] = d * d; __syncthreads();
    for (int st = 128; st > 0; st >>= 1) { if (c < st) red[c] += red[c + st]; __syncthreads(); }
    float rstd = rsqrtf(red[0] * (1.f / CC) + EPSV);
    hout[(size_t)t * CC + c] = (__bf16)(d * rstd * n2g[c] + n2b[c]);
    x2[(size_t)t * CC + c] = xv;
}

// ---------------------------------------------------------------------------
extern "C" void kernel_launch(void* const* d_in, const int* in_sizes, int n_in,
                              void* d_out, int out_size, void* d_ws, size_t ws_size,
                              hipStream_t stream) {
    const float* x       = (const float*)d_in[0];
    const float* n1g     = (const float*)d_in[1];
    const float* n1b     = (const float*)d_in[2];
    const float* qkv_w   = (const float*)d_in[3];
    const float* qkv_b   = (const float*)d_in[4];
    const float* rpb     = (const float*)d_in[5];
    const float* proj_w  = (const float*)d_in[6];
    const float* proj_b  = (const float*)d_in[7];
    const float* n2g     = (const float*)d_in[8];
    const float* n2b     = (const float*)d_in[9];
    const float* fc1_w   = (const float*)d_in[10];
    const float* fc1_b   = (const float*)d_in[11];
    const float* fc2_w   = (const float*)d_in[12];
    const float* fc2_b   = (const float*)d_in[13];
    const float* cabn_g  = (const float*)d_in[14];
    const float* cabn_b  = (const float*)d_in[15];
    const float* cab1_w  = (const float*)d_in[16];
    const float* cab1_b  = (const float*)d_in[17];
    const float* cab2_w  = (const float*)d_in[18];
    const float* cab2_b  = (const float*)d_in[19];
    const float* dw_w    = (const float*)d_in[20];
    const float* dw_b    = (const float*)d_in[21];
    float* out = (float*)d_out;

    char* ws = (char*)d_ws;
    size_t off = 0;
    auto alloc = [&](size_t bytes) -> void* {
        void* p = ws + off;
        off += (bytes + 255) & ~(size_t)255;
        return p;
    };
    __bf16* wq  = (__bf16*)alloc((size_t)3 * CC * CC * 2);
    __bf16* wp  = (__bf16*)alloc((size_t)CC * CC * 2);
    __bf16* w1  = (__bf16*)alloc((size_t)MLPH * CC * 2);
    __bf16* w2  = (__bf16*)alloc((size_t)CC * MLPH * 2);
    float*  gap = (float*)alloc((size_t)BB * CC * 4);
    float*  gmp = (float*)alloc((size_t)BB * CC * 4);
    float*  ca  = (float*)alloc((size_t)BB * CC * 4);
    // segA: xw (steps 2-3), reused as hln (steps 9-10)
    __bf16* xw  = (__bf16*)alloc((size_t)MTOT * CC * 2);
    __bf16* hln = xw;
    // segB: qkv (201MB) + attn (67MB) contiguous; reused whole as h2 (268MB)
    __bf16* qkvb = (__bf16*)alloc((size_t)MTOT * 3 * CC * 2);
    __bf16* attn = (__bf16*)alloc((size_t)MTOT * CC * 2);
    __bf16* h2   = qkvb;
    float*  x1   = (float*)alloc((size_t)MTOT * CC * 4);
    float*  yln  = (float*)alloc((size_t)MTOT * CC * 4);
    float*  x2   = (float*)alloc((size_t)MTOT * CC * 4);
    (void)ws_size; (void)in_sizes; (void)n_in; (void)out_size;

    // 1) weight conversion
    f2bf_kernel<<<(3*CC*CC + 255)/256, 256, 0, stream>>>(qkv_w, wq, 3*CC*CC);
    f2bf_kernel<<<(CC*CC   + 255)/256, 256, 0, stream>>>(proj_w, wp, CC*CC);
    f2bf_kernel<<<(MLPH*CC + 255)/256, 256, 0, stream>>>(fc1_w, w1, MLPH*CC);
    f2bf_kernel<<<(CC*MLPH + 255)/256, 256, 0, stream>>>(fc2_w, w2, CC*MLPH);

    // 2) LN1 + shift + window partition -> bf16
    ln_kernel<true, true><<<MTOT/8, 256, 0, stream>>>(x, n1g, n1b, nullptr, xw);

    // 3) QKV GEMM: [MTOT,256] x [768,256]^T -> bf16
    gemm_kernel<0><<<(MTOT/128)*((3*CC)/64), 256, 0, stream>>>(xw, wq, qkv_b, nullptr, nullptr, qkvb, MTOT, 3*CC, CC);

    // 4) windowed attention (bias + mask + softmax), WMMA QK^T and PV
    attn_kernel<<<BB * NWIN * 2, 256, 0, stream>>>(qkvb, rpb, attn);

    // 5) proj GEMM + window-reverse/roll permute + residual(x) -> x1 fp32
    gemm_kernel<3><<<(MTOT/128)*(CC/64), 256, 0, stream>>>(attn, wp, proj_b, x, x1, nullptr, MTOT, CC, CC);

    // 6) CAB LayerNorm -> yln fp32
    ln_kernel<false, false><<<MTOT/8, 256, 0, stream>>>(x1, cabn_g, cabn_b, yln, nullptr);

    // 7) GAP/GMP per (b,c)
    gapgmp_kernel<<<BB, 256, 0, stream>>>(yln, gap, gmp);

    // 8) SE MLP -> channel attention
    se_kernel<<<BB, 256, 0, stream>>>(gap, gmp, cab1_w, cab1_b, cab2_w, cab2_b, ca);

    // 9) CAB scale + depthwise conv + residual, fused with LN2 -> x2, hln(bf16)
    cab_fuse_kernel<<<MTOT, 256, 0, stream>>>(x1, yln, ca, dw_w, dw_b, n2g, n2b, x2, hln);

    // 10) FC1 GEMM + exact GELU -> h2 bf16
    gemm_kernel<1><<<(MTOT/128)*(MLPH/64), 256, 0, stream>>>(hln, w1, fc1_b, nullptr, nullptr, h2, MTOT, MLPH, CC);

    // 11) FC2 GEMM + residual(x2) -> out fp32
    gemm_kernel<2><<<(MTOT/128)*(CC/64), 256, 0, stream>>>(h2, w2, fc2_b, x2, out, nullptr, MTOT, CC, MLPH);
}
